// GCN_Regression_37641093382384
// MI455X (gfx1250) — matrix operand order repaired and checked by reading, not compiled, not run on hardware
//
#include <hip/hip_runtime.h>
#include <hip/hip_bf16.h>

// ---------------------------------------------------------------------------
// Types for CDNA5 WMMA (wave32)
// ---------------------------------------------------------------------------
typedef _Float16 half_t;
typedef __attribute__((ext_vector_type(16))) _Float16 v16h;
typedef __attribute__((ext_vector_type(8)))  _Float16 v8h;
typedef __attribute__((ext_vector_type(8)))  float    v8f;

#define T_LEN 512
#define B_SZ  64
#define NSAMP (T_LEN * B_SZ)   // 32768 (B*T) samples
#define NUM_J 17
#define EPS   1e-5f

// ---------------------------------------------------------------------------
// WMMA fragment helpers (v_wmma_f32_16x16x32_f16)
//
// A fragment: row-major (M x K) source, tile 16x32.
//   lane: m = lane&15, kh = lane>>4 ; slot s -> k = kh*16 + s
//   => one contiguous 16-half (32B) load per lane.
// B fragment: B = W^T where W is row-major (N x K); tile covers K=32, N=16.
//   lane: n = lane&15, kh = lane>>4 ; slot s -> k = kh*16 + s
//   => one contiguous 16-half (32B) load from row n of W.
// C/D: lane n = lane&15 ; vgpr v -> m = v + 8*(lane>>4).
// ---------------------------------------------------------------------------
__device__ __forceinline__ v16h frag_a(const half_t* base, int ld) {
  const int lane = threadIdx.x & 31;
  const int m  = lane & 15;
  const int kh = lane >> 4;
  return *(const v16h*)(base + m * ld + (kh << 4));
}

__device__ __forceinline__ v16h frag_b_wt(const half_t* w, int ld) {
  const int lane = threadIdx.x & 31;
  const int n  = lane & 15;
  const int kh = lane >> 4;
  return *(const v16h*)(w + n * ld + (kh << 4));
}

__device__ __forceinline__ v8f wmma16(v16h a, v16h b, v8f c) {
  // 8 args: (neg_a, A, neg_b, B, c_mod, C, reuse_a, reuse_b)
  return __builtin_amdgcn_wmma_f32_16x16x32_f16(false, a, false, b, (short)0, c,
                                                false, false);
}

// ---------------------------------------------------------------------------
// Async copy helpers (GLOBAL_LOAD_ASYNC_TO_LDS_B128, tracked by ASYNCcnt).
// Generic LDS addresses carry the LDS byte offset in their low 32 bits
// (ISA 10.2), which is what VDST wants.
// ---------------------------------------------------------------------------
__device__ __forceinline__ void async_ld_b128(void* lds_dst, const void* gsrc) {
  const unsigned int loff = (unsigned int)(uintptr_t)lds_dst;
  asm volatile("global_load_async_to_lds_b128 %0, %1, off"
               :: "v"(loff), "v"(gsrc)
               : "memory");
}
__device__ __forceinline__ void wait_asynccnt0() {
  asm volatile("s_wait_asynccnt 0x0" ::: "memory");
}

// ===========================================================================
// Kernel 1: data_bn + 3 fused GCN layers (support GEMM via WMMA, A-mix + BN
// + ReLU in VALU) + joint-mean pooling.  4 samples (68 joint-rows -> 5
// m-tiles) per workgroup; everything stays in LDS between layers.
// ===========================================================================
#define GCN_SAMP 4
#define GCN_ROWS (GCN_SAMP * NUM_J)   // 68
#define GCN_MT   5                    // 80 padded rows
#define GCN_LDA  128                  // halves, input cols padded

struct GcnLayer { const float *W, *b, *bn_g, *bn_b, *bn_m, *bn_v; };
struct GcnArgs {
  const float* x;       // (B,T,17,6)
  const float* A;       // (17,17)
  const float *db_g, *db_b, *db_m, *db_v;
  GcnLayer L[3];
  half_t* seq_out;      // (T,B,256) f16
};

#define GCN_SMEM (80*GCN_LDA*2 /*Ah*/ + 80*256*4 /*Sup*/ + 256*128*2 /*Wh*/ + 1216 /*Aadj*/)

__global__ void __launch_bounds__(256) gcn_fused(GcnArgs args) {
  extern __shared__ char smem[];
  half_t* Ah   = (half_t*)smem;                               // 80 x 128 f16
  float*  Sup  = (float*)(smem + 80*GCN_LDA*2);               // 80 x 256 f32
  half_t* Wh   = (half_t*)((char*)Sup + 80*256*4);            // 256 x 128 f16
  float*  Aadj = (float*)((char*)Wh + 256*128*2);             // 17 x 17 f32

  const int tid  = threadIdx.x;
  const int wave = tid >> 5;
  const int lane = tid & 31;
  const int n0s  = blockIdx.x * GCN_SAMP;

  for (int i = tid; i < NUM_J*NUM_J; i += 256) Aadj[i] = args.A[i];

  // Stage layer-0 input with eval-BN applied; zero all padding.
  for (int i = tid; i < 80 * GCN_LDA; i += 256) {
    const int r = i >> 7, c = i & 127;
    float val = 0.f;
    if (r < GCN_ROWS && c < 6) {
      const int s = r / NUM_J, j = r % NUM_J;
      const int n = n0s + s;
      const float xv = args.x[((size_t)n * NUM_J + j) * 6 + c];
      const float sc = args.db_g[c] * rsqrtf(args.db_v[c] + EPS);
      val = (xv - args.db_m[c]) * sc + args.db_b[c];
    }
    Ah[r * GCN_LDA + c] = (half_t)val;
  }

  const int CinP[3] = {32, 64, 128};   // K padded to WMMA chunk
  const int Cout[3] = {64, 128, 256};

  for (int layer = 0; layer < 3; ++layer) {
    __syncthreads();
    const int Kp = CinP[layer], Co = Cout[layer];
    const int Kw = (layer == 0) ? 6 : Kp;     // real K in the weight tensor
    const GcnLayer Lp = args.L[layer];

    // Stage W (Co x Kw) -> LDS f16 (Co x Kp), zero K padding.
    for (int i = tid; i < Co * Kp; i += 256) {
      const int n = i / Kp, k = i - n * Kp;
      Wh[i] = (half_t)((k < Kw) ? Lp.W[n * Kw + k] : 0.f);
    }
    __syncthreads();

    // support = h @ W^T + b   via WMMA tiles
    const int ntiles = Co >> 4, kch = Kp >> 5;
    for (int tile = wave; tile < GCN_MT * ntiles; tile += 8) {
      const int mt = tile / ntiles, nt = tile - mt * ntiles;
      const half_t* Ab = Ah + mt * 16 * GCN_LDA;
      const half_t* Bb = Wh + nt * 16 * Kp;
      v8f acc = {};
      for (int kc = 0; kc < kch; ++kc) {
        v16h a = frag_a(Ab + kc * 32, GCN_LDA);
        v16h b = frag_b_wt(Bb + kc * 32, Kp);
        acc = wmma16(a, b, acc);
      }
      const int nl = lane & 15, mh = lane >> 4;
      const float bias = Lp.b[nt * 16 + nl];
#pragma unroll
      for (int v = 0; v < 8; ++v)
        Sup[(mt * 16 + mh * 8 + v) * Co + nt * 16 + nl] = acc[v] + bias;
    }
    __syncthreads();

    // h = relu(BN(A_hat @ support)) ; last layer: BN only + joint-mean pool
    const bool last = (layer == 2);
    for (int p = tid; p < GCN_SAMP * Co; p += 256) {
      const int s = p / Co, c = p - s * Co;
      const float sc = Lp.bn_g[c] * rsqrtf(Lp.bn_v[c] + EPS);
      const float sh = Lp.bn_b[c] - Lp.bn_m[c] * sc;
      const float* Sb = Sup + (size_t)(s * NUM_J) * Co + c;
      float pool = 0.f;
      for (int j = 0; j < NUM_J; ++j) {
        float acc = 0.f;
        const float* ar = Aadj + j * NUM_J;
#pragma unroll
        for (int k = 0; k < NUM_J; ++k) acc += ar[k] * Sb[k * Co];
        float val = acc * sc + sh;
        if (!last) {
          val = fmaxf(val, 0.f);
          Ah[(s * NUM_J + j) * GCN_LDA + c] = (half_t)val;
        } else {
          pool += val;
        }
      }
      if (last) {
        const int n = n0s + s;
        const int bi = n >> 9, t = n & (T_LEN - 1);
        args.seq_out[((size_t)t * B_SZ + bi) * 256 + c] =
            (half_t)(pool * (1.f / (float)NUM_J));
      }
    }
  }
}

// ===========================================================================
// Kernel 2: GRU input-gate GEMM  gi = seq @ Wih^T + bih
//   seq: (32768, 256) f16   Wih: (384, 256) f32   gi: (32768, 384) f16
//   WG tile: 64 x 128, K = 256 (8 chunks). A frags loaded straight from
//   global (already f16); Wih block staged to LDS f16.
// ===========================================================================
#define GI_SMEM (128*256*2)

__global__ void __launch_bounds__(256)
gemm_gi(const half_t* __restrict__ seq, const float* __restrict__ Wih,
        const float* __restrict__ bih, half_t* __restrict__ gi) {
  extern __shared__ char smem[];
  half_t* Wl = (half_t*)smem;                      // 128 x 256 f16
  const int tid  = threadIdx.x;
  const int wave = tid >> 5;
  const int lane = tid & 31;
  const int m0   = blockIdx.x * 64;
  const int n0b  = blockIdx.y * 128;

  for (int i = tid; i < 128 * 256; i += 256)
    Wl[i] = (half_t)Wih[(size_t)(n0b + (i >> 8)) * 256 + (i & 255)];
  __builtin_prefetch(seq + (size_t)m0 * 256, 0, 1);
  __syncthreads();

  for (int tile = wave; tile < 32; tile += 8) {     // 4 m-tiles x 8 n-tiles
    const int mt = tile >> 3, nt = tile & 7;
    const half_t* Ab = seq + (size_t)(m0 + mt * 16) * 256;
    const half_t* Bb = Wl + nt * 16 * 256;
    v8f acc = {};
#pragma unroll
    for (int kc = 0; kc < 8; ++kc) {
      v16h a = frag_a(Ab + kc * 32, 256);
      v16h b = frag_b_wt(Bb + kc * 32, 256);
      acc = wmma16(a, b, acc);
    }
    const int nl = lane & 15, mh = lane >> 4;
    const int gcol = n0b + nt * 16 + nl;
    const float bias = bih[gcol];
#pragma unroll
    for (int v = 0; v < 8; ++v)
      gi[(size_t)(m0 + mt * 16 + mh * 8 + v) * 384 + gcol] =
          (half_t)(acc[v] + bias);
  }
}

// ===========================================================================
// Kernel 3: GRU recurrence. grid.x = 2 (dir 0 = forward, 1 = backward).
// One persistent workgroup per direction; Whh (f16) + hidden state resident
// in LDS. Per step: async-prefetch next step's gi slice (48KB) into the
// ping-pong LDS buffer (GLOBAL_LOAD_ASYNC_TO_LDS_B128 / ASYNCcnt) while the
// current step's gh = h @ Whh^T WMMAs run; then gate math entirely out of LDS.
// LDS: 96K Whh + 2x48K gi + 48K gh(f16) + 32K h(f32) + 16K h(f16) = 288KB.
// ===========================================================================
struct GruDir { const half_t* gi; const float* Whh; const float* bhh; };

#define GI_STEP_BYTES (B_SZ * 384 * 2)            // 49152
#define GRU_SMEM (384*128*2 + 2*GI_STEP_BYTES + 64*384*2 + 64*128*4 + 64*128*2)

__global__ void __launch_bounds__(256)
gru_recur(GruDir dirf, GruDir dirb, half_t* __restrict__ seq_out,
          float* __restrict__ hTf, float* __restrict__ hTb) {
  extern __shared__ char smem[];
  half_t* Wl   = (half_t*)smem;                              // 384 x 128 f16
  half_t* giB0 = (half_t*)(smem + 384*128*2);                // 64 x 384 f16
  half_t* giB1 = (half_t*)(smem + 384*128*2 + GI_STEP_BYTES);
  half_t* GHh  = (half_t*)(smem + 384*128*2 + 2*GI_STEP_BYTES); // 64x384 f16
  float*  Hf   = (float*)((char*)GHh + 64*384*2);            // 64 x 128 f32
  half_t* Hh   = (half_t*)((char*)Hf + 64*128*4);            // 64 x 128 f16

  const int dir = blockIdx.x;
  const GruDir P = dir ? dirb : dirf;
  float* hT = dir ? hTb : hTf;
  const int colOff = dir ? 128 : 0;
  const int tid  = threadIdx.x;
  const int wave = tid >> 5;
  const int lane = tid & 31;

  for (int i = tid; i < 384 * 128; i += 256) Wl[i] = (half_t)P.Whh[i];
  for (int i = tid; i < 64 * 128; i += 256) { Hf[i] = 0.f; Hh[i] = (half_t)0.f; }

  // Prologue: async-load gi slice for the first step into buffer 0.
  {
    const int tt0 = dir ? (T_LEN - 1) : 0;
    const char* g = (const char*)P.gi + (size_t)tt0 * GI_STEP_BYTES;
    char* l = (char*)giB0;
    for (int q = tid; q < GI_STEP_BYTES / 16; q += 256)
      async_ld_b128(l + q * 16, g + q * 16);
  }

  for (int t = 0; t < T_LEN; ++t) {
    const int tt = dir ? (T_LEN - 1 - t) : t;
    half_t* giCur = (t & 1) ? giB1 : giB0;

    wait_asynccnt0();       // current gi buffer resident in LDS
    __syncthreads();

    // Kick off async prefetch of the NEXT step's gi into the other buffer.
    if (t + 1 < T_LEN) {
      const int tn = dir ? (T_LEN - 2 - t) : (t + 1);
      const char* g = (const char*)P.gi + (size_t)tn * GI_STEP_BYTES;
      char* l = (char*)((t & 1) ? giB0 : giB1);
      for (int q = tid; q < GI_STEP_BYTES / 16; q += 256)
        async_ld_b128(l + q * 16, g + q * 16);
    }

    // gh = h @ Whh^T + bhh : 4 m-tiles x 24 n-tiles, K = 128
    for (int tile = wave; tile < 96; tile += 8) {
      const int mt = tile / 24, nt = tile - mt * 24;
      const half_t* Ab = Hh + mt * 16 * 128;
      const half_t* Bb = Wl + nt * 16 * 128;
      v8f acc = {};
#pragma unroll
      for (int kc = 0; kc < 4; ++kc) {
        v16h a = frag_a(Ab + kc * 32, 128);
        v16h b = frag_b_wt(Bb + kc * 32, 128);
        acc = wmma16(a, b, acc);
      }
      const int nl = lane & 15, mh = lane >> 4;
      const float bias = P.bhh[nt * 16 + nl];
#pragma unroll
      for (int v = 0; v < 8; ++v)
        GHh[(mt * 16 + mh * 8 + v) * 384 + nt * 16 + nl] =
            (half_t)(acc[v] + bias);
    }
    __syncthreads();

    // Gate math: everything from LDS.
    for (int e = tid; e < 64 * 128; e += 256) {
      const int br = e >> 7, c = e & 127;
      const float ir = (float)giCur[br * 384 + c];
      const float iz = (float)giCur[br * 384 + 128 + c];
      const float in = (float)giCur[br * 384 + 256 + c];
      const float hr = (float)GHh[br * 384 + c];
      const float hz = (float)GHh[br * 384 + 128 + c];
      const float hn = (float)GHh[br * 384 + 256 + c];
      const float r = 1.f / (1.f + __expf(-(ir + hr)));
      const float z = 1.f / (1.f + __expf(-(iz + hz)));
      const float n = tanhf(in + r * hn);
      const float h = (1.f - z) * n + z * Hf[e];
      Hf[e] = h;
      Hh[e] = (half_t)h;
      seq_out[((size_t)tt * B_SZ + br) * 256 + colOff + c] = (half_t)h;
      if (t == T_LEN - 1) hT[e] = h;
    }
  }
}

// ===========================================================================
// Kernel 4: head MLP + LayerNorm + output transform (tiny; VALU only)
// ===========================================================================
#define HEAD_SMEM (64*288*4 + 64*128*4)

__global__ void __launch_bounds__(256)
head_k(const float* hTf, const float* hTb, const float* ex_embed,
       const int* exid, const float* W1, const float* b1, const float* ln_g,
       const float* ln_b, const float* W2, const float* b2, const float* W3,
       const float* b3, float* out) {
  extern __shared__ char smem[];
  float* feat = (float*)smem;            // 64 x 288
  float* zbuf = feat + 64 * 288;         // 64 x 128
  const int tid = threadIdx.x;

  for (int i = tid; i < 64 * 288; i += 256) {
    const int r = i / 288, c = i - r * 288;
    float v;
    if (c < 128)      v = hTf[r * 128 + c];
    else if (c < 256) v = hTb[r * 128 + (c - 128)];
    else              v = ex_embed[exid[r] * 32 + (c - 256)];
    feat[i] = v;
  }
  __syncthreads();
  for (int e = tid; e < 64 * 128; e += 256) {
    const int r = e >> 7, c = e & 127;
    float acc = b1[c];
    const float* fr = feat + r * 288;
    const float* wr = W1 + c * 288;
    for (int k = 0; k < 288; ++k) acc += fr[k] * wr[k];
    zbuf[e] = acc;
  }
  __syncthreads();
  if (tid < 64) {
    const float* zr = zbuf + tid * 128;
    float mu = 0.f;
    for (int k = 0; k < 128; ++k) mu += zr[k];
    mu *= (1.f / 128.f);
    float var = 0.f;
    for (int k = 0; k < 128; ++k) { const float d = zr[k] - mu; var += d * d; }
    const float rs = rsqrtf(var * (1.f / 128.f) + EPS);
    float z2[64];
    for (int c2 = 0; c2 < 64; ++c2) {
      float acc = b2[c2];
      const float* w2r = W2 + c2 * 128;
      for (int k = 0; k < 128; ++k) {
        float v = (zr[k] - mu) * rs * ln_g[k] + ln_b[k];
        acc += w2r[k] * fmaxf(v, 0.f);
      }
      z2[c2] = fmaxf(acc, 0.f);
    }
    float o = b3[0];
    for (int c2 = 0; c2 < 64; ++c2) o += W3[c2] * z2[c2];
    out[tid] = 3.f + 2.f * tanhf(o);
  }
}

// ===========================================================================
// Host launch
// ===========================================================================
extern "C" void kernel_launch(void* const* d_in, const int* in_sizes, int n_in,
                              void* d_out, int out_size, void* d_ws,
                              size_t ws_size, hipStream_t stream) {
  (void)in_sizes; (void)n_in; (void)out_size; (void)ws_size;
  // Input order: x, exercise_id, A_hat, then params leaves (JAX pytree:
  // alphabetical dict keys at every level).
  const float* x      = (const float*)d_in[0];
  const int*   exid   = (const int*)d_in[1];
  const float* A_hat  = (const float*)d_in[2];
  const float* dbn_b  = (const float*)d_in[3];
  const float* dbn_g  = (const float*)d_in[4];
  const float* dbn_m  = (const float*)d_in[5];
  const float* dbn_v  = (const float*)d_in[6];
  const float* ex_emb = (const float*)d_in[7];
  // gcn[i] @ 8+6i: W, b, bn_b, bn_g, bn_m, bn_v
  // gru[l] @ 26+8l: Whh_b, Whh_f, Wih_b, Wih_f, bhh_b, bhh_f, bih_b, bih_f
  const float* hW1   = (const float*)d_in[42];
  const float* hW2   = (const float*)d_in[43];
  const float* hW3   = (const float*)d_in[44];
  const float* hb1   = (const float*)d_in[45];
  const float* hb2   = (const float*)d_in[46];
  const float* hb3   = (const float*)d_in[47];
  const float* hln_b = (const float*)d_in[48];
  const float* hln_g = (const float*)d_in[49];

  char* ws = (char*)d_ws;
  const size_t SEQ_B = (size_t)NSAMP * 256 * sizeof(half_t);
  const size_t GI_B  = (size_t)NSAMP * 384 * sizeof(half_t);
  half_t* seqA = (half_t*)ws;
  half_t* seqB = (half_t*)(ws + SEQ_B);
  half_t* gif  = (half_t*)(ws + 2 * SEQ_B);
  half_t* gib  = (half_t*)(ws + 2 * SEQ_B + GI_B);
  float*  hTf  = (float*)(ws + 2 * SEQ_B + 2 * GI_B);
  float*  hTb  = hTf + 64 * 128;

  GcnArgs ga;
  ga.x = x; ga.A = A_hat;
  ga.db_g = dbn_g; ga.db_b = dbn_b; ga.db_m = dbn_m; ga.db_v = dbn_v;
  for (int i = 0; i < 3; ++i) {
    ga.L[i].W    = (const float*)d_in[8 + 6 * i + 0];
    ga.L[i].b    = (const float*)d_in[8 + 6 * i + 1];
    ga.L[i].bn_b = (const float*)d_in[8 + 6 * i + 2];
    ga.L[i].bn_g = (const float*)d_in[8 + 6 * i + 3];
    ga.L[i].bn_m = (const float*)d_in[8 + 6 * i + 4];
    ga.L[i].bn_v = (const float*)d_in[8 + 6 * i + 5];
  }
  ga.seq_out = seqA;
  gcn_fused<<<dim3(NSAMP / GCN_SAMP), dim3(256), GCN_SMEM, stream>>>(ga);

  half_t* seqIn = seqA;
  half_t* seqOut = seqB;
  for (int l = 0; l < 2; ++l) {
    const float* Whh_b = (const float*)d_in[26 + 8 * l + 0];
    const float* Whh_f = (const float*)d_in[26 + 8 * l + 1];
    const float* Wih_b = (const float*)d_in[26 + 8 * l + 2];
    const float* Wih_f = (const float*)d_in[26 + 8 * l + 3];
    const float* bhh_b = (const float*)d_in[26 + 8 * l + 4];
    const float* bhh_f = (const float*)d_in[26 + 8 * l + 5];
    const float* bih_b = (const float*)d_in[26 + 8 * l + 6];
    const float* bih_f = (const float*)d_in[26 + 8 * l + 7];

    gemm_gi<<<dim3(NSAMP / 64, 3), dim3(256), GI_SMEM, stream>>>(seqIn, Wih_f, bih_f, gif);
    gemm_gi<<<dim3(NSAMP / 64, 3), dim3(256), GI_SMEM, stream>>>(seqIn, Wih_b, bih_b, gib);

    GruDir df{gif, Whh_f, bhh_f};
    GruDir db{gib, Whh_b, bhh_b};
    gru_recur<<<dim3(2), dim3(256), GRU_SMEM, stream>>>(df, db, seqOut, hTf, hTb);

    half_t* tmp = seqIn; seqIn = seqOut; seqOut = tmp;
  }

  head_k<<<dim3(1), dim3(256), HEAD_SMEM, stream>>>(
      hTf, hTb, ex_emb, exid, hW1, hb1, hln_g, hln_b, hW2, hb2, hW3, hb3,
      (float*)d_out);
}
